// BaselineAttention_6751688589994
// MI455X (gfx1250) — matrix operand order repaired and checked by reading, not compiled, not run on hardware
//
#include <hip/hip_runtime.h>
#include <hip/hip_bf16.h>
#include <cstdint>

// MHA block: B=4, S=2048, HID=1024, NH=16, HD=64, causal.
// Matmuls via v_wmma_f32_16x16x32_bf16; K/V tiles staged by the Tensor Data
// Mover with double buffering (copy for tile kt+1 overlaps compute on kt).

typedef __attribute__((ext_vector_type(16))) __bf16 v16bf;
typedef __attribute__((ext_vector_type(8)))  float  v8f;
typedef __attribute__((ext_vector_type(4)))  unsigned int v4u;
typedef __attribute__((ext_vector_type(8)))  int v8i;
typedef __attribute__((ext_vector_type(4)))  int v4i;

#define BATCH 4
#define SEQ   2048
#define HID_  1024
#define NH_   16
#define HD_   64
#define SCALE_ 0.125f   // 1/sqrt(64)

#if defined(__has_builtin)
#if __has_builtin(__builtin_amdgcn_tensor_load_to_lds)
#define HAVE_TDM 1
#endif
#endif

#if __has_include(<hip/amd_detail/amd_gfx1250_TDM.h>) || (__clang_major__ >= 23)
#define TDM_6ARG 1
#endif

union FragBF {  uint4 u[2]; v16bf v; };
union Half8  {  uint4 u;    __bf16 h[8]; };

__device__ inline v16bf load_frag(const __bf16* p0, const __bf16* p1) {
    FragBF f;
    f.u[0] = *(const uint4*)p0;
    f.u[1] = *(const uint4*)p1;
    return f.v;
}

__device__ inline v8f bwmma(v16bf a, v16bf b, v8f c) {
    // D = A(16x32 bf16) x B(32x16 bf16) + C(16x16 f32)
    return __builtin_amdgcn_wmma_f32_16x16x32_bf16(
        /*neg_a=*/false, a, /*neg_b=*/false, b,
        /*c_mod=*/(short)0, c, /*reuse_a=*/false, /*reuse_b=*/false);
}

#ifdef HAVE_TDM
// TDM 2D tile load: tile_h rows x tile_w bf16 elements, row stride in elements.
// D# layout per CDNA5 ISA ch.8 (group0 128b, group1 256b, groups 2/3 zero = 2D).
__device__ inline void tdm_load_2d(uint32_t lds_addr, const void* gaddr,
                                   uint32_t tile_w, uint32_t tile_h,
                                   uint32_t row_stride_elems) {
    const uint64_t ga = (uint64_t)(uintptr_t)gaddr;
    v4u g0;
    g0[0] = 1u;                                   // count=1 (valid), user mode
    g0[1] = lds_addr;                             // LDS byte address (bits 63:32)
    g0[2] = (uint32_t)(ga & 0xFFFFFFFFu);         // global_addr[31:0]
    g0[3] = (uint32_t)((ga >> 32) & 0x01FFFFFFu)  // global_addr[56:32]
          | (2u << 30);                           // type=2 ("image")
    v8i g1;
    g1[0] = (int)(1u << 16);                      // wg_mask=0, data_size=1 (2B)
    g1[1] = (int)(tile_w << 16);                  // tensor_dim0[15:0] @ bits 63:48
    g1[2] = (int)(tile_h << 16);                  // dim0 hi=0; tensor_dim1[15:0] @ 95:80
    g1[3] = (int)(tile_w << 16);                  // dim1 hi=0; tile_dim0 @ 127:112
    g1[4] = (int)(tile_h);                        // tile_dim1 @ 143:128; tile_dim2=0
    g1[5] = (int)(row_stride_elems);              // tensor_dim0_stride[31:0] @ 191:160
    g1[6] = 0;                                    // stride hi / dim1_stride lo
    g1[7] = 0;
    v4i z = {0, 0, 0, 0};
#ifdef TDM_6ARG
    v8i z8 = {0, 0, 0, 0, 0, 0, 0, 0};
    __builtin_amdgcn_tensor_load_to_lds(g0, g1, z, z, z8, 0);
#else
    __builtin_amdgcn_tensor_load_to_lds(g0, g1, z, z, 0);
#endif
}
#endif

// ---------------------------------------------------------------- convert
__global__ void cvt_f32_bf16(const float* __restrict__ in, __bf16* __restrict__ out, int n) {
    int i = blockIdx.x * blockDim.x + threadIdx.x;
    if (i < n) out[i] = (__bf16)in[i];
}

// ---------------------------------------------------------------- GEMM: Out = X @ W^T
// X: MxK bf16 row-major, W: NxK bf16 row-major (B columns = W rows, k-contiguous)
// 256 thr = 8 waves; wave tile 16 M x 64 N (A-fragment reused across 4 WMMAs,
// 4 independent accumulator chains); block tile 64 M x 128 N.
template<bool OUT_F32>
__global__ __launch_bounds__(256) void gemm_xwT(const __bf16* __restrict__ X,
                                                const __bf16* __restrict__ W,
                                                void* __restrict__ OutP,
                                                int M, int N, int K) {
    const int lane = threadIdx.x & 31;
    const int wave = threadIdx.x >> 5;
    const int ln = lane & 15;
    const int hf = lane >> 4;
    const int m0 = blockIdx.x * 64  + (wave & 3) * 16;
    const int n0 = blockIdx.y * 128 + (wave >> 2) * 64;

    const __bf16* xrow = X + (size_t)(m0 + ln) * K;        // A row m = ln
    const __bf16* wr0  = W + (size_t)(n0 +  0 + ln) * K;   // B col n = ln (+16j)
    const __bf16* wr1  = W + (size_t)(n0 + 16 + ln) * K;
    const __bf16* wr2  = W + (size_t)(n0 + 32 + ln) * K;
    const __bf16* wr3  = W + (size_t)(n0 + 48 + ln) * K;
    const int ao = hf * 16;       // A lane halves: K 0..15 / 16..31
    const int bo = hf * 16;       // B lane halves: K 0..15 / 16..31

    v8f c0 = {}, c1 = {}, c2 = {}, c3 = {};
    for (int k0 = 0; k0 < K; k0 += 32) {
        if (k0 + 64 < K) {
            __builtin_prefetch(xrow + k0 + 64, 0, 0);
            __builtin_prefetch(wr0 + k0 + 64, 0, 0);
            __builtin_prefetch(wr2 + k0 + 64, 0, 0);
        }
        v16bf a  = load_frag(xrow + k0 + ao, xrow + k0 + ao + 8);
        v16bf b0 = load_frag(wr0 + k0 + bo, wr0 + k0 + bo + 8);
        v16bf b1 = load_frag(wr1 + k0 + bo, wr1 + k0 + bo + 8);
        v16bf b2 = load_frag(wr2 + k0 + bo, wr2 + k0 + bo + 8);
        v16bf b3 = load_frag(wr3 + k0 + bo, wr3 + k0 + bo + 8);
        c0 = bwmma(a, b0, c0);
        c1 = bwmma(a, b1, c1);
        c2 = bwmma(a, b2, c2);
        c3 = bwmma(a, b3, c3);
    }
    // C layout: VGPR r -> row m0 + hf*8 + r, col n0 + 16j + ln
    const int row = m0 + hf * 8;
    const int col = n0 + ln;
    if (OUT_F32) {
        float* Out = (float*)OutP;
        #pragma unroll
        for (int r = 0; r < 8; ++r) {
            const size_t base = (size_t)(row + r) * N + col;
            Out[base]      = c0[r];
            Out[base + 16] = c1[r];
            Out[base + 32] = c2[r];
            Out[base + 48] = c3[r];
        }
    } else {
        __bf16* Out = (__bf16*)OutP;
        #pragma unroll
        for (int r = 0; r < 8; ++r) {
            const size_t base = (size_t)(row + r) * N + col;
            Out[base]      = (__bf16)c0[r];
            Out[base + 16] = (__bf16)c1[r];
            Out[base + 32] = (__bf16)c2[r];
            Out[base + 48] = (__bf16)c3[r];
        }
    }
}

// ---------------------------------------------------------------- RoPE (in place, bf16)
// out[j]    = x[j]*cos - x[2j+1]*sin ;  out[j+32] = x[j+32]*cos + x[2j]*sin
__global__ void rope_kernel(__bf16* __restrict__ QK, int total /* B*S*NH */) {
    int i = blockIdx.x * blockDim.x + threadIdx.x;
    if (i >= total) return;
    const int h  = i % NH_;
    const int bs = i / NH_;
    const int s  = bs % SEQ;
    __bf16* vec = QK + (size_t)bs * HID_ + h * HD_;
    float x[HD_];
    #pragma unroll
    for (int j = 0; j < HD_; ++j) x[j] = (float)vec[j];
    const float t = (float)s;
    #pragma unroll
    for (int j = 0; j < HD_ / 2; ++j) {
        // inv_freq = 10000^(-j/32) = exp(-j * ln(10000)/32)
        float invf = __expf(-(float)j * (9.210340371976184f / 32.0f));
        float ang = t * invf;
        float cs = __cosf(ang), sn = __sinf(ang);
        float o0 = x[j]      * cs - x[2 * j + 1] * sn;
        float o1 = x[j + 32] * cs + x[2 * j]     * sn;
        vec[j]      = (__bf16)o0;
        vec[j + 32] = (__bf16)o1;
    }
}

// ---------------------------------------------------------------- flash attention
// grid (S/128, B*NH); 256 thr = 8 waves; wave owns 16 q rows; key tiles of 32.
// TDM double buffering: the DMA for tile kt+1 runs while tile kt is computed.
__global__ __launch_bounds__(256) void flash_attn(const __bf16* __restrict__ Q,
                                                  const __bf16* __restrict__ K,
                                                  const __bf16* __restrict__ V,
                                                  __bf16* __restrict__ Ctx) {
#ifdef HAVE_TDM
    __shared__ __bf16 sK[2][32][64];    // [buf][key][d]
    __shared__ __bf16 sV[2][32][64];    // [buf][key][d] raw V tile (TDM target)
#else
    __shared__ __bf16 sK[1][32][64];
#endif
    __shared__ __bf16 sVt[64][32];      // [d][key] transposed for P*V B-frags
    __shared__ __bf16 sP[8][16][32];    // per-wave P staging [q][key]

    const int b = blockIdx.y / NH_;
    const int h = blockIdx.y % NH_;
    const int q0 = blockIdx.x * 128;
    const int lane = threadIdx.x & 31;
    const int wave = threadIdx.x >> 5;
    const int ln = lane & 15;
    const int hf = lane >> 4;

    // Q A-fragments (row m = ln), d chunks 0..31 and 32..63
    v16bf qa0, qa1;
    {
        const __bf16* qrow =
            Q + (size_t)(b * SEQ + q0 + wave * 16 + ln) * HID_ + h * HD_;
        const int ao = hf * 16;
        qa0 = load_frag(qrow + ao,      qrow + ao + 8);
        qa1 = load_frag(qrow + 32 + ao, qrow + 32 + ao + 8);
    }

    v8f acc0 = {}, acc1 = {}, acc2 = {}, acc3 = {};   // ctx 16 x 64 (4 n-tiles of d)
    float mrow[8], lrow[8];
    #pragma unroll
    for (int r = 0; r < 8; ++r) { mrow[r] = -1e30f; lrow[r] = 0.0f; }

    const size_t bh_base = (size_t)(b * SEQ) * HID_ + h * HD_;
    const int ntiles = q0 / 32 + 4;    // causal bound for this block's 128 q rows

#ifdef HAVE_TDM
    if (threadIdx.x < 32) {            // prologue: tile 0 into buffer 0
        tdm_load_2d((uint32_t)(uintptr_t)&sK[0][0][0], K + bh_base, HD_, 32, HID_);
        tdm_load_2d((uint32_t)(uintptr_t)&sV[0][0][0], V + bh_base, HD_, 32, HID_);
    }
#endif
    for (int kt = 0; kt < ntiles; ++kt) {
#ifdef HAVE_TDM
        const int cur = kt & 1;
        if (threadIdx.x < 32)
            __builtin_amdgcn_s_wait_tensorcnt(0);   // buffer `cur` complete
        __syncthreads();   // A: publish buf cur; buf 1-cur reads (iter kt-1) done
        if ((threadIdx.x < 32) && (kt + 1 < ntiles)) {
            const size_t nxt = bh_base + (size_t)(kt + 1) * 32 * HID_;
            tdm_load_2d((uint32_t)(uintptr_t)&sK[1 - cur][0][0], K + nxt, HD_, 32, HID_);
            tdm_load_2d((uint32_t)(uintptr_t)&sV[1 - cur][0][0], V + nxt, HD_, 32, HID_);
        }
        {   // LDS->LDS transpose of V (8 bf16 per thread)
            const int key = threadIdx.x >> 3;
            const int ds  = (threadIdx.x & 7) * 8;
            Half8 vv; vv.u = *(const uint4*)&sV[cur][key][ds];
            #pragma unroll
            for (int j = 0; j < 8; ++j) sVt[ds + j][key] = vv.h[j];
        }
#else
        const int cur = 0;
        __syncthreads();
        {   // fallback: cooperative global staging, 8 bf16 per thread
            const int key = threadIdx.x >> 3;
            const int ds  = (threadIdx.x & 7) * 8;
            const size_t rowbase = bh_base + (size_t)(kt * 32 + key) * HID_ + ds;
            *(uint4*)&sK[0][key][ds] = *(const uint4*)(K + rowbase);
            Half8 vv; vv.u = *(const uint4*)(V + rowbase);
            #pragma unroll
            for (int j = 0; j < 8; ++j) sVt[ds + j][key] = vv.h[j];
        }
#endif
        __syncthreads();   // B: sVt (and sK tile) ready for compute

        // scores: 16 q x 32 keys = two 16x16 frags, accumulate over d (2 x K=32)
        v8f s0 = {}, s1 = {};
        #pragma unroll
        for (int dc = 0; dc < 2; ++dc) {
            const int kb = dc * 32 + hf * 16;
            v16bf kf0 = load_frag(&sK[cur][ln][kb],      &sK[cur][ln][kb + 8]);
            v16bf kf1 = load_frag(&sK[cur][16 + ln][kb], &sK[cur][16 + ln][kb + 8]);
            v16bf qa = dc ? qa1 : qa0;
            s0 = bwmma(qa, kf0, s0);
            s1 = bwmma(qa, kf1, s1);
        }

        // online softmax; row m = hf*8 + r lives in this half's 16 lanes
        const int qbase = q0 + wave * 16 + hf * 8;
        #pragma unroll
        for (int r = 0; r < 8; ++r) {
            const int qr  = qbase + r;
            const int kg  = kt * 32 + ln;
            float v0 = s0[r] * SCALE_ + ((kg      > qr) ? -1e9f : 0.0f);
            float v1 = s1[r] * SCALE_ + ((kg + 16 > qr) ? -1e9f : 0.0f);
            float tm = fmaxf(v0, v1);
            #pragma unroll
            for (int off = 8; off > 0; off >>= 1)
                tm = fmaxf(tm, __shfl_xor(tm, off, 16));
            float nm = fmaxf(mrow[r], tm);
            float fs = __expf(mrow[r] - nm);
            float p0 = __expf(v0 - nm);
            float p1 = __expf(v1 - nm);
            float rs = p0 + p1;
            #pragma unroll
            for (int off = 8; off > 0; off >>= 1)
                rs += __shfl_xor(rs, off, 16);
            lrow[r] = lrow[r] * fs + rs;
            mrow[r] = nm;
            acc0[r] *= fs; acc1[r] *= fs; acc2[r] *= fs; acc3[r] *= fs;
            sP[wave][hf * 8 + r][ln]      = (__bf16)p0;
            sP[wave][hf * 8 + r][16 + ln] = (__bf16)p1;
        }
        // wave-local: make P stores visible to this wave's A-fragment reloads
        asm volatile("s_wait_dscnt 0" ::: "memory");

        // ctx += P(16x32) x V(32x64): A from sP, B from sVt (keys contiguous)
        {
            const int o = hf * 16;
            v16bf pa  = load_frag(&sP[wave][ln][o], &sP[wave][ln][o + 8]);
            v16bf vb0 = load_frag(&sVt[ln][hf * 16],      &sVt[ln][hf * 16 + 8]);
            v16bf vb1 = load_frag(&sVt[16 + ln][hf * 16], &sVt[16 + ln][hf * 16 + 8]);
            v16bf vb2 = load_frag(&sVt[32 + ln][hf * 16], &sVt[32 + ln][hf * 16 + 8]);
            v16bf vb3 = load_frag(&sVt[48 + ln][hf * 16], &sVt[48 + ln][hf * 16 + 8]);
            acc0 = bwmma(pa, vb0, acc0);
            acc1 = bwmma(pa, vb1, acc1);
            acc2 = bwmma(pa, vb2, acc2);
            acc3 = bwmma(pa, vb3, acc3);
        }
    }

    // epilogue: divide by l, write ctx back as [b, q, h*64 + d] bf16
    const int row = q0 + wave * 16 + hf * 8;
    const int col = h * HD_ + ln;
    #pragma unroll
    for (int r = 0; r < 8; ++r) {
        const float inv = 1.0f / lrow[r];
        const size_t base = (size_t)(b * SEQ + row + r) * HID_ + col;
        Ctx[base]      = (__bf16)(acc0[r] * inv);
        Ctx[base + 16] = (__bf16)(acc1[r] * inv);
        Ctx[base + 32] = (__bf16)(acc2[r] * inv);
        Ctx[base + 48] = (__bf16)(acc3[r] * inv);
    }
}

// ---------------------------------------------------------------- host
extern "C" void kernel_launch(void* const* d_in, const int* in_sizes, int n_in,
                              void* d_out, int out_size, void* d_ws, size_t ws_size,
                              hipStream_t stream) {
    const float* X  = (const float*)d_in[0];
    // d_in[1] = attention_mask: causal mask is applied analytically in flash_attn.
    const float* Wq = (const float*)d_in[2];
    const float* Wk = (const float*)d_in[3];
    const float* Wv = (const float*)d_in[4];
    const float* Wo = (const float*)d_in[5];

    const size_t MS = (size_t)BATCH * SEQ;          // 8192 rows
    const size_t XN = MS * HID_;                    // 8M elems
    const size_t WN = (size_t)HID_ * HID_;          // 1M elems

    char* p = (char*)d_ws;
    __bf16* Xb  = (__bf16*)p; p += XN * 2;
    __bf16* Wqb = (__bf16*)p; p += WN * 2;
    __bf16* Wkb = (__bf16*)p; p += WN * 2;
    __bf16* Wvb = (__bf16*)p; p += WN * 2;
    __bf16* Wob = (__bf16*)p; p += WN * 2;
    __bf16* Qb  = (__bf16*)p; p += XN * 2;
    __bf16* Kb  = (__bf16*)p; p += XN * 2;
    __bf16* Vb  = (__bf16*)p; p += XN * 2;
    __bf16* Cb  = (__bf16*)p; p += XN * 2;          // total ~88 MB

    // 1) fp32 -> bf16
    cvt_f32_bf16<<<(int)(XN / 256), 256, 0, stream>>>(X,  Xb,  (int)XN);
    cvt_f32_bf16<<<(int)(WN / 256), 256, 0, stream>>>(Wq, Wqb, (int)WN);
    cvt_f32_bf16<<<(int)(WN / 256), 256, 0, stream>>>(Wk, Wkb, (int)WN);
    cvt_f32_bf16<<<(int)(WN / 256), 256, 0, stream>>>(Wv, Wvb, (int)WN);
    cvt_f32_bf16<<<(int)(WN / 256), 256, 0, stream>>>(Wo, Wob, (int)WN);

    // 2) QKV projections (WMMA bf16), block tile 64 M x 128 N
    dim3 gg((unsigned)(MS / 64), HID_ / 128);
    gemm_xwT<false><<<gg, 256, 0, stream>>>(Xb, Wqb, Qb, (int)MS, HID_, HID_);
    gemm_xwT<false><<<gg, 256, 0, stream>>>(Xb, Wkb, Kb, (int)MS, HID_, HID_);
    gemm_xwT<false><<<gg, 256, 0, stream>>>(Xb, Wvb, Vb, (int)MS, HID_, HID_);

    // 3) RoPE on Q and K (in place)
    const int heads = (int)(MS * NH_);
    rope_kernel<<<heads / 256, 256, 0, stream>>>(Qb, heads);
    rope_kernel<<<heads / 256, 256, 0, stream>>>(Kb, heads);

    // 4) causal flash attention (double-buffered TDM staging)
    dim3 ga(SEQ / 128, BATCH * NH_);
    flash_attn<<<ga, 256, 0, stream>>>(Qb, Kb, Vb, Cb);

    // 5) output projection -> f32 d_out
    gemm_xwT<true><<<gg, 256, 0, stream>>>(Cb, Wob, d_out, (int)MS, HID_, HID_);
}
